// TRN2V2_81046032876221
// MI455X (gfx1250) — matrix operand order repaired and checked by reading, not compiled, not run on hardware
//
#include <hip/hip_runtime.h>
#include <hip/hip_bf16.h>

// ---------------- problem dims ----------------
#define BDIM   128
#define TDIM   64
#define FDIM   2048
#define HDIM   2048
#define CDIM   22
#define DSTEPS 8
#define G4     8192      // 4*F == 4*H (gate width)
#define KF     2048      // K for h@Whh^T and h@trW^T
#define CPAD   32        // C padded to WMMA K multiple
#define KFUSE  2080      // (F + C) = 2070 padded to multiple of 32

typedef __attribute__((ext_vector_type(16))) __bf16 v16bf;
typedef __attribute__((ext_vector_type(8)))  __bf16 v8bf;
typedef __attribute__((ext_vector_type(8)))  float  v8f;

__device__ __forceinline__ unsigned short f2bf(float f) {
  unsigned int u = __float_as_uint(f);
  unsigned int r = (u + 0x7FFFu + ((u >> 16) & 1u)) >> 16;  // RNE
  return (unsigned short)r;
}

// ---------------------------------------------------------------------------
// GEMM: C[M=128, n<Nreal] (=/+=) A[128,K]bf16 (row-major, lda) @ W[N,K]bf16^T
// Block = 256 threads = 8 waves. Wave w owns M-tile rows [16w,16w+16).
// Block b owns N-strip of NT*16 columns; NT accumulators per wave, the A
// fragment is reused across all NT WMMAs. NT is compile-time: the k-loop is
// branch-free and software-pipelined (double-buffered A/B fragments) so the
// WMMAs for step k overlap the global loads for step k+1 (partial loadcnt
// waits instead of full drains). bf16 WMMA 16x16x32, fp32 accumulate.
// ---------------------------------------------------------------------------
template <int NT>
__global__ __launch_bounds__(256) void gemm_bf16_wmma(
    const unsigned short* __restrict__ A, int lda,
    const unsigned short* __restrict__ W, int ldw,
    float* __restrict__ C, int ldc,
    const float* __restrict__ bias,
    int K, int Nreal, int addC)
{
  const int lane = threadIdx.x & 31;
  const int wave = threadIdx.x >> 5;
  const int half = lane >> 4;      // 0: lanes 0-15, 1: lanes 16-31
  const int l16  = lane & 15;
  const int m0   = wave * 16;
  const int nblk = blockIdx.x * (NT * 16);

  // accumulator init: bias (first GEMM of a sum) and/or prior C (accumulate)
  v8f acc[NT];
  #pragma unroll
  for (int nt = 0; nt < NT; ++nt) {
    const int n = nblk + nt * 16 + l16;
    float vb = (bias && n < Nreal) ? bias[n] : 0.0f;
    #pragma unroll
    for (int r = 0; r < 8; ++r) {
      float v = vb;
      if (addC && n < Nreal)
        v += C[(size_t)(m0 + 8 * half + r) * (size_t)ldc + n];
      acc[nt][r] = v;
    }
  }

  // A fragment (16x32): lane m0+l16 holds
  // K = {kb..kb+15}, kb = k0 + 16*half -> two b128 loads.
  const unsigned short* Arow = A + (size_t)(m0 + l16) * (size_t)lda + half * 16;
  // B fragment (32x16): lane holds column n, 16 contiguous K of W row n.
  const unsigned short* Wrow = W + (size_t)(nblk + l16) * (size_t)ldw + half * 16;

  // ---- software pipeline: prologue loads for k0 = 0 ----
  v8bf  alo = *(const v8bf*)(Arow);
  v8bf  ahi = *(const v8bf*)(Arow + 8);
  v16bf bcur[NT];
  #pragma unroll
  for (int nt = 0; nt < NT; ++nt)
    bcur[nt] = *(const v16bf*)(Wrow + (size_t)nt * 16 * (size_t)ldw);

  #pragma unroll 2
  for (int k0 = 0; k0 < K - 32; k0 += 32) {
    // issue next-step loads while current WMMAs execute
    v8bf  alo2 = *(const v8bf*)(Arow + k0 + 32);
    v8bf  ahi2 = *(const v8bf*)(Arow + k0 + 40);
    v16bf bnxt[NT];
    #pragma unroll
    for (int nt = 0; nt < NT; ++nt)
      bnxt[nt] = *(const v16bf*)(Wrow + (size_t)nt * 16 * (size_t)ldw + k0 + 32);

    v16bf a;
    #pragma unroll
    for (int i = 0; i < 8; ++i) { a[i] = alo[i]; a[i + 8] = ahi[i]; }
    #pragma unroll
    for (int nt = 0; nt < NT; ++nt)
      acc[nt] = __builtin_amdgcn_wmma_f32_16x16x32_bf16(
          false, a, false, bcur[nt], (short)0, acc[nt], false, false);

    alo = alo2; ahi = ahi2;
    #pragma unroll
    for (int nt = 0; nt < NT; ++nt) bcur[nt] = bnxt[nt];
  }

  // ---- epilogue: last k-step ----
  {
    v16bf a;
    #pragma unroll
    for (int i = 0; i < 8; ++i) { a[i] = alo[i]; a[i + 8] = ahi[i]; }
    #pragma unroll
    for (int nt = 0; nt < NT; ++nt)
      acc[nt] = __builtin_amdgcn_wmma_f32_16x16x32_bf16(
          false, a, false, bcur[nt], (short)0, acc[nt], false, false);
  }

  // store D: M = m0 + r + 8*half, N = nblk + nt*16 + l16
  #pragma unroll
  for (int nt = 0; nt < NT; ++nt) {
    const int n = nblk + nt * 16 + l16;
    if (n < Nreal) {
      #pragma unroll
      for (int r = 0; r < 8; ++r)
        C[(size_t)(m0 + 8 * half + r) * (size_t)ldc + n] = acc[nt][r];
    }
  }
}

// ------------------- LSTM pointwise (i,f,g,o gate order) -------------------
__global__ void lstm_pw(const float* __restrict__ gates,
                        float* __restrict__ c,
                        unsigned short* __restrict__ hbf)
{
  int idx = blockIdx.x * blockDim.x + threadIdx.x;
  if (idx >= BDIM * HDIM) return;
  int m = idx >> 11;         // / 2048
  int j = idx & 2047;
  const float* g = gates + (size_t)m * G4;
  float gi = g[j];
  float gf = g[HDIM + j];
  float gg = g[2 * HDIM + j];
  float go = g[3 * HDIM + j];
  float si = 1.0f / (1.0f + __expf(-gi));
  float sf = 1.0f / (1.0f + __expf(-gf));
  float so = 1.0f / (1.0f + __expf(-go));
  float c2 = sf * c[idx] + si * tanhf(gg);
  float h2 = so * tanhf(c2);
  c[idx]   = c2;
  hbf[idx] = f2bf(h2);
}

// --------- fp32 -> bf16 weight convert with zero-pad of extra cols ---------
__global__ void cvt_pad(const float* __restrict__ src,
                        unsigned short* __restrict__ dst,
                        int scols, int dcols, int total)
{
  int idx = blockIdx.x * blockDim.x + threadIdx.x;
  if (idx >= total) return;
  int r = idx / dcols;
  int j = idx - r * dcols;
  dst[idx] = (j < scols) ? f2bf(src[(size_t)r * scols + j]) : (unsigned short)0;
}

__global__ void addvec(const float* __restrict__ a, const float* __restrict__ b,
                       float* __restrict__ o, int n)
{
  int i = blockIdx.x * blockDim.x + threadIdx.x;
  if (i < n) o[i] = a[i] + b[i];
}

// zero encoder state (h bf16, c fp32)
__global__ void init_state(unsigned short* __restrict__ ehbf,
                           float* __restrict__ ec)
{
  int idx = blockIdx.x * blockDim.x + threadIdx.x;
  if (idx >= BDIM * HDIM) return;
  ehbf[idx] = 0;
  ec[idx]   = 0.0f;
}

// per encoder step: xt -> fused[:, :F] (bf16), decoder h0 = xt (bf16),
// decoder c = 0, fut = 0
__global__ void cast_x_kernel(const float* __restrict__ xt, int rstride,
                              unsigned short* __restrict__ fused,
                              unsigned short* __restrict__ hdec,
                              float* __restrict__ cdec,
                              unsigned short* __restrict__ futbf)
{
  int idx = blockIdx.x * blockDim.x + threadIdx.x;
  if (idx >= BDIM * FDIM) return;
  int m = idx >> 11;
  int j = idx & 2047;
  unsigned short bv = f2bf(xt[(size_t)m * rstride + j]);
  fused[(size_t)m * KFUSE + j] = bv;
  hdec[idx] = bv;
  cdec[idx] = 0.0f;
  if (j < CPAD) futbf[m * CPAD + j] = 0;
}

// decoder scores slice (fp32, in d_out) -> bf16 padded fut; optionally also
// fill fused[:, F:F+CPAD] (covers zero-pad of cols 2070..2079)
__global__ void cast_fut(const float* __restrict__ src, int rstride,
                         unsigned short* __restrict__ futbf,
                         unsigned short* __restrict__ fused)
{
  int idx = blockIdx.x * blockDim.x + threadIdx.x;
  if (idx >= BDIM * CPAD) return;
  int m = idx >> 5;
  int j = idx & 31;
  unsigned short v = (j < CDIM) ? f2bf(src[(size_t)m * rstride + j])
                                : (unsigned short)0;
  futbf[idx] = v;
  if (fused) fused[(size_t)m * KFUSE + FDIM + j] = v;
}

// ---------------------------------------------------------------------------
extern "C" void kernel_launch(void* const* d_in, const int* in_sizes, int n_in,
                              void* d_out, int out_size, void* d_ws, size_t ws_size,
                              hipStream_t stream)
{
  (void)in_sizes; (void)n_in; (void)out_size; (void)ws_size;

  const float* x      = (const float*)d_in[0];
  const float* decWih = (const float*)d_in[1];   // [8192, 22]
  const float* decWhh = (const float*)d_in[2];   // [8192, 2048]
  const float* decbih = (const float*)d_in[3];
  const float* decbhh = (const float*)d_in[4];
  const float* trW    = (const float*)d_in[5];   // [22, 2048]
  const float* trb    = (const float*)d_in[6];
  const float* encWih = (const float*)d_in[7];   // [8192, 2070]
  const float* encWhh = (const float*)d_in[8];   // [8192, 2048]
  const float* encbih = (const float*)d_in[9];
  const float* encbhh = (const float*)d_in[10];
  const float* clsW   = (const float*)d_in[11];  // [22, 2048]
  const float* clsb   = (const float*)d_in[12];

  float* out     = (float*)d_out;
  float* enc_out = out;                                   // [B, T, C]
  float* dec_out = out + (size_t)BDIM * TDIM * CDIM;      // [B, T, D, C]

  // ---- workspace carve-up (≈104 MB; bf16 weights stay L2-resident) ----
  char* ws = (char*)d_ws;
  size_t off = 0;
  auto take = [&](size_t bytes) -> void* {
    void* p = ws + off;
    off = (off + bytes + 255) & ~(size_t)255;
    return p;
  };
  unsigned short* wDecIh = (unsigned short*)take((size_t)G4 * CPAD * 2);
  unsigned short* wDecHh = (unsigned short*)take((size_t)G4 * KF * 2);
  unsigned short* wTr    = (unsigned short*)take((size_t)CDIM * KF * 2);
  unsigned short* wEncIh = (unsigned short*)take((size_t)G4 * KFUSE * 2);
  unsigned short* wEncHh = (unsigned short*)take((size_t)G4 * KF * 2);
  unsigned short* wCls   = (unsigned short*)take((size_t)CDIM * KF * 2);
  float*          bDec   = (float*)take((size_t)G4 * 4);
  float*          bEnc   = (float*)take((size_t)G4 * 4);
  float*          gates  = (float*)take((size_t)BDIM * G4 * 4);
  float*          cdec   = (float*)take((size_t)BDIM * HDIM * 4);
  float*          ec     = (float*)take((size_t)BDIM * HDIM * 4);
  unsigned short* hdec   = (unsigned short*)take((size_t)BDIM * HDIM * 2);
  unsigned short* ehbf   = (unsigned short*)take((size_t)BDIM * HDIM * 2);
  unsigned short* fused  = (unsigned short*)take((size_t)BDIM * KFUSE * 2);
  unsigned short* futbf  = (unsigned short*)take((size_t)BDIM * CPAD * 2);

  const int TPB = 256;
  auto nb = [](size_t n) { return (unsigned)((n + 255) / 256); };

  // ---- once per call: weights -> bf16 (zero-padded K), bias sums, state ----
  cvt_pad<<<nb((size_t)G4 * CPAD), TPB, 0, stream>>>(decWih, wDecIh, CDIM, CPAD, G4 * CPAD);
  cvt_pad<<<nb((size_t)G4 * KF),   TPB, 0, stream>>>(decWhh, wDecHh, KF,   KF,   G4 * KF);
  cvt_pad<<<nb((size_t)CDIM * KF), TPB, 0, stream>>>(trW,    wTr,    KF,   KF,   CDIM * KF);
  cvt_pad<<<nb((size_t)G4 * KFUSE),TPB, 0, stream>>>(encWih, wEncIh, FDIM + CDIM, KFUSE, G4 * KFUSE);
  cvt_pad<<<nb((size_t)G4 * KF),   TPB, 0, stream>>>(encWhh, wEncHh, KF,   KF,   G4 * KF);
  cvt_pad<<<nb((size_t)CDIM * KF), TPB, 0, stream>>>(clsW,   wCls,   KF,   KF,   CDIM * KF);
  addvec<<<nb(G4), TPB, 0, stream>>>(decbih, decbhh, bDec, G4);
  addvec<<<nb(G4), TPB, 0, stream>>>(encbih, encbhh, bEnc, G4);
  init_state<<<nb((size_t)BDIM * HDIM), TPB, 0, stream>>>(ehbf, ec);

  const int dec_ldc = TDIM * DSTEPS * CDIM;  // dec_scores row stride (per b)
  const int enc_ldc = TDIM * CDIM;           // enc_scores row stride (per b)

  for (int t = 0; t < TDIM; ++t) {
    // xt -> fused[:, :F], decoder h0 = xt, decoder c = 0, fut = 0
    cast_x_kernel<<<nb((size_t)BDIM * FDIM), TPB, 0, stream>>>(
        x + (size_t)t * FDIM, TDIM * FDIM, fused, hdec, cdec, futbf);

    for (int d = 0; d < DSTEPS; ++d) {
      // gates = fut @ decWih^T + (bih + bhh)       (N=8192, K=32)
      gemm_bf16_wmma<4><<<G4 / 64, TPB, 0, stream>>>(
          futbf, CPAD, wDecIh, CPAD, gates, G4, bDec, CPAD, G4, 0);
      // gates += h @ decWhh^T   (dominant 128x8192x2048 GEMM)
      gemm_bf16_wmma<4><<<G4 / 64, TPB, 0, stream>>>(
          hdec, KF, wDecHh, KF, gates, G4, nullptr, KF, G4, 1);
      // LSTM cell update -> cdec (fp32), hdec (bf16)
      lstm_pw<<<nb((size_t)BDIM * HDIM), TPB, 0, stream>>>(gates, cdec, hdec);
      // fut = h2 @ trW^T + trb  -> straight into dec_scores[b, t, d, :]
      float* dsc = dec_out + (size_t)t * DSTEPS * CDIM + (size_t)d * CDIM;
      gemm_bf16_wmma<2><<<1, TPB, 0, stream>>>(
          hdec, KF, wTr, KF, dsc, dec_ldc, trb, KF, CDIM, 0);
      // bf16-pad fut for next step; last step also fills fused[:, F:]
      cast_fut<<<nb((size_t)BDIM * CPAD), TPB, 0, stream>>>(
          dsc, dec_ldc, futbf, (d == DSTEPS - 1) ? fused : nullptr);
    }

    // egates = [xt | fut] @ encWih^T + (bih + bhh)   (K=2080)
    gemm_bf16_wmma<4><<<G4 / 64, TPB, 0, stream>>>(
        fused, KFUSE, wEncIh, KFUSE, gates, G4, bEnc, KFUSE, G4, 0);
    // egates += eh @ encWhh^T
    gemm_bf16_wmma<4><<<G4 / 64, TPB, 0, stream>>>(
        ehbf, KF, wEncHh, KF, gates, G4, nullptr, KF, G4, 1);
    lstm_pw<<<nb((size_t)BDIM * HDIM), TPB, 0, stream>>>(gates, ec, ehbf);
    // enc_scores[b, t, :] = eh2 @ clsW^T + clsb
    gemm_bf16_wmma<2><<<1, TPB, 0, stream>>>(
        ehbf, KF, wCls, KF, enc_out + (size_t)t * CDIM, enc_ldc,
        clsb, KF, CDIM, 0);
  }
}
